// DeformableAttention_24060406792537
// MI455X (gfx1250) — matrix operand order repaired and checked by reading, not compiled, not run on hardware
//
#include <hip/hip_runtime.h>
#include <cstdint>

typedef __attribute__((ext_vector_type(16))) _Float16 v16h;
typedef __attribute__((ext_vector_type(8)))  float    v8f;

#define LQ      22223
#define CDIM    256
#define NHEADS  8
#define HDIM    32

// ---------------------------------------------------------------------------
// CDNA5 async global->LDS copy (GLOBAL_LOAD_ASYNC_TO_LDS_B128, ASYNCcnt),
// guarded so we fall back to load+ds_store if the builtins are absent.
// ROCm 7.2 clang-22 prototype (from its own diagnostic):
//   void __builtin_amdgcn_global_load_async_to_lds_b128(
//       __attribute__((address_space(1))) int_v4* src,   // printed "__device__"
//       __attribute__((address_space(3))) int_v4* dst,   // LDS
//       int imm_offset, int cpol);
// ---------------------------------------------------------------------------
#if defined(__has_builtin)
#  if __has_builtin(__builtin_amdgcn_global_load_async_to_lds_b128) && \
      __has_builtin(__builtin_amdgcn_s_wait_asynccnt)
#    define HAVE_ASYNC_LDS 1
#  endif
#endif
#ifndef HAVE_ASYNC_LDS
#  define HAVE_ASYNC_LDS 0
#endif

typedef int v4i_gcc __attribute__((vector_size(16)));

__device__ __forceinline__ void copy16_to_lds(const void* g, void* l) {
#if HAVE_ASYNC_LDS
    __builtin_amdgcn_global_load_async_to_lds_b128(
        (__attribute__((address_space(1))) v4i_gcc*)(uintptr_t)g,
        (__attribute__((address_space(3))) v4i_gcc*)(uintptr_t)l, 0, 0);
#else
    *(uint4*)l = *(const uint4*)g;
#endif
}
__device__ __forceinline__ void lds_copy_wait() {
#if HAVE_ASYNC_LDS
    __builtin_amdgcn_s_wait_asynccnt(0);
#endif
}

// ---------------------------------------------------------------------------
// fp32 -> f16 cast (elementwise)
// ---------------------------------------------------------------------------
__global__ void cast_f16_kernel(const float* __restrict__ in,
                                _Float16* __restrict__ out, int n) {
    int i = blockIdx.x * blockDim.x + threadIdx.x;
    if (i < n) out[i] = (_Float16)in[i];
}

// fp32 (K x N, row-major) -> f16 transposed (N x K, row-major)
__global__ void cast_transpose_f16_kernel(const float* __restrict__ in,
                                          _Float16* __restrict__ out,
                                          int K, int N) {
    int i = blockIdx.x * blockDim.x + threadIdx.x;
    if (i >= K * N) return;
    int k = i / N, n = i % N;
    out[(size_t)n * K + k] = (_Float16)in[i];
}

// ---------------------------------------------------------------------------
// WMMA GEMM: C[M,N] = A[M,K] * B[K,N] + bias,  A f16 (MxK), Bt f16 (NxK).
// Block = 256 threads = 8 waves. BM=128, BN=64, BK=32.
// Wave w computes rows [w*16, w*16+16) x 4 column tiles of 16.
// ---------------------------------------------------------------------------
template <int STORE_F16>
__global__ __launch_bounds__(256)
void gemm_wmma_f16(const _Float16* __restrict__ A,
                   const _Float16* __restrict__ Bt,
                   const float* __restrict__ bias,
                   float* __restrict__ Cf,
                   _Float16* __restrict__ Ch,
                   int M, int N, int K) {
    __shared__ __align__(16) _Float16 As[128 * 32];
    __shared__ __align__(16) _Float16 Bs[64 * 32];

    const int tid  = threadIdx.x;
    const int wave = tid >> 5;
    const int lane = tid & 31;
    const int mblk = blockIdx.x * 128;
    const int nblk = blockIdx.y * 64;

    v8f acc[4] = {};

    // A fragment addressing (16-bit A 16x32 layout): lane<16 -> K 0..7 & 16..23,
    // lane>=16 -> K 8..15 & 24..31; row = lane%16.
    const int rowA = wave * 16 + (lane & 15);
    const int k0a  = (lane < 16) ? 0 : 8;
    // B fragment addressing (16-bit B 32x16): col = lane%16,
    // lane<16 -> K 0..7 & 16..23, lane>=16 -> K 8..15 & 24..31 (matches A).
    const int colB = lane & 15;
    const int k0b  = (lane < 16) ? 0 : 8;

    // staging coordinates (per thread)
    const int ar = tid >> 1;                 // 0..127
    const int ac = (tid & 1) * 16;           // halves
    int agr = mblk + ar; if (agr >= M) agr = M - 1;
    const int br = tid >> 2;                 // 0..63
    const int bc = (tid & 3) * 8;            // halves
    const int bgn = nblk + br;               // N is a multiple of 64

    for (int kt = 0; kt < K; kt += 32) {
        // ---- stage A tile: 128 rows x 32 halves (8 KB); 32 B per thread ----
        copy16_to_lds(A + (size_t)agr * K + kt + ac,      &As[ar * 32 + ac]);
        copy16_to_lds(A + (size_t)agr * K + kt + ac + 8,  &As[ar * 32 + ac + 8]);
        // ---- stage B tile: 64 cols x 32 halves (4 KB); 16 B per thread ----
        copy16_to_lds(Bt + (size_t)bgn * K + kt + bc,     &Bs[br * 32 + bc]);
        if (kt + 32 < K)                      // prefetch next A tile into L2
            __builtin_prefetch(A + (size_t)agr * K + kt + 32 + ac, 0, 1);
        lds_copy_wait();
        __syncthreads();

        v16h a, b0, b1, b2, b3;
        {
            const _Float16* p = &As[rowA * 32 + k0a];
            ((uint4*)&a)[0] = *(const uint4*)p;          // K k0a..k0a+7
            ((uint4*)&a)[1] = *(const uint4*)(p + 16);   // K k0a+16..k0a+23
        }
        {
            const _Float16* p = &Bs[colB * 32 + k0b];
            ((uint4*)&b0)[0] = *(const uint4*)p;
            ((uint4*)&b0)[1] = *(const uint4*)(p + 16);
            ((uint4*)&b1)[0] = *(const uint4*)(p + 16 * 32);
            ((uint4*)&b1)[1] = *(const uint4*)(p + 16 * 32 + 16);
            ((uint4*)&b2)[0] = *(const uint4*)(p + 32 * 32);
            ((uint4*)&b2)[1] = *(const uint4*)(p + 32 * 32 + 16);
            ((uint4*)&b3)[0] = *(const uint4*)(p + 48 * 32);
            ((uint4*)&b3)[1] = *(const uint4*)(p + 48 * 32 + 16);
        }
        // independent B registers -> the four WMMAs can issue back-to-back
        acc[0] = __builtin_amdgcn_wmma_f32_16x16x32_f16(
            false, a, false, b0, (short)0, acc[0], false, false);
        acc[1] = __builtin_amdgcn_wmma_f32_16x16x32_f16(
            false, a, false, b1, (short)0, acc[1], false, false);
        acc[2] = __builtin_amdgcn_wmma_f32_16x16x32_f16(
            false, a, false, b2, (short)0, acc[2], false, false);
        acc[3] = __builtin_amdgcn_wmma_f32_16x16x32_f16(
            false, a, false, b3, (short)0, acc[3], false, false);
        __syncthreads();
    }

    // ---- store: C/D layout -> lane<16: rows r+0..7 ; lane>=16: rows r+8..15
    const int rbase = mblk + wave * 16 + ((lane < 16) ? 0 : 8);
    const int cl    = lane & 15;
#pragma unroll
    for (int ct = 0; ct < 4; ++ct) {
        int col  = nblk + ct * 16 + cl;
        float bv = bias[col];
#pragma unroll
        for (int r = 0; r < 8; ++r) {
            int row = rbase + r;
            if (row < M) {
                float v = acc[ct][r] + bv;
                if (STORE_F16) Ch[(size_t)row * N + col] = (_Float16)v;
                else           Cf[(size_t)row * N + col] = v;
            }
        }
    }
}

// ---------------------------------------------------------------------------
// Sampling: one wave per (query, head); lane = channel (HDIM==warpSize==32).
// softmax over 16 logits, 16 bilinear gathers from f16 value maps, weighted sum.
// ---------------------------------------------------------------------------
__global__ __launch_bounds__(256)
void ms_deform_sample(const float* __restrict__ off,     // LQ x 256
                      const float* __restrict__ attnl,   // LQ x 128
                      const float* __restrict__ refp,    // LQ x 4 x 4
                      const _Float16* __restrict__ v,    // LV x 256 (f16)
                      _Float16* __restrict__ outh) {     // LQ x 256
    const int q    = blockIdx.x;
    const int h    = threadIdx.x >> 5;
    const int lane = threadIdx.x & 31;

    // softmax over the 16 (level,point) logits of this head (wave-uniform)
    float lg[16];
    const float* ap = attnl + (size_t)q * 128 + h * 16;
    float m = -1e30f;
#pragma unroll
    for (int i = 0; i < 16; ++i) { lg[i] = ap[i]; m = fmaxf(m, lg[i]); }
    float s = 0.f;
#pragma unroll
    for (int i = 0; i < 16; ++i) { lg[i] = __expf(lg[i] - m); s += lg[i]; }
    const float inv = 1.f / s;

    const int Hs[4]     = {100, 50, 25, 13};
    const int Ws[4]     = {167, 84, 42, 21};
    const int starts[4] = {0, 16700, 20900, 21950};

    float acc = 0.f;
    const float* op = off + (size_t)q * 256 + h * 32;   // (l*4+p)*2 indexing
    const float* rp = refp + (size_t)q * 16;

#pragma unroll
    for (int l = 0; l < 4; ++l) {
        const float cx = rp[l * 4 + 0], cy = rp[l * 4 + 1];
        const float rw = rp[l * 4 + 2], rh = rp[l * 4 + 3];
        const int   H = Hs[l], W = Ws[l];
        const float fW = (float)W, fH = (float)H;
        const size_t base = (size_t)starts[l] * 256 + h * 32 + lane;
#pragma unroll
        for (int p = 0; p < 4; ++p) {
            float ox = op[(l * 4 + p) * 2 + 0];
            float oy = op[(l * 4 + p) * 2 + 1];
            // loc = c + off/4 * wh * 0.5 ; x = loc*W - 0.5 (grid math folded)
            float x = (cx + ox * 0.125f * rw) * fW - 0.5f;
            float y = (cy + oy * 0.125f * rh) * fH - 0.5f;
            float x0f = floorf(x), y0f = floorf(y);
            int   x0 = (int)x0f, y0 = (int)y0f;
            float wx1 = x - x0f, wy1 = y - y0f;
            float wx0 = 1.f - wx1, wy0 = 1.f - wy1;
            float vs = 0.f;
#pragma unroll
            for (int dy = 0; dy < 2; ++dy) {
                int iy = y0 + dy;
                if (iy < 0 || iy >= H) continue;      // wave-uniform branch
                float wy = dy ? wy1 : wy0;
#pragma unroll
                for (int dx = 0; dx < 2; ++dx) {
                    int ix = x0 + dx;
                    if (ix < 0 || ix >= W) continue;
                    float wx = dx ? wx1 : wx0;
                    float val = (float)v[base + (size_t)(iy * W + ix) * 256];
                    vs += wy * wx * val;
                }
            }
            acc += lg[l * 4 + p] * inv * vs;
        }
    }
    outh[(size_t)q * 256 + h * 32 + lane] = (_Float16)acc;
}

// ---------------------------------------------------------------------------
extern "C" void kernel_launch(void* const* d_in, const int* in_sizes, int n_in,
                              void* d_out, int out_size, void* d_ws, size_t ws_size,
                              hipStream_t stream) {
    const float* query  = (const float*)d_in[0];
    const float* refp   = (const float*)d_in[1];
    const float* value  = (const float*)d_in[2];
    const float* W_off  = (const float*)d_in[3];
    const float* b_off  = (const float*)d_in[4];
    const float* W_attn = (const float*)d_in[5];
    const float* b_attn = (const float*)d_in[6];
    const float* W_val  = (const float*)d_in[7];
    const float* b_val  = (const float*)d_in[8];
    const float* W_out  = (const float*)d_in[9];
    const float* b_out  = (const float*)d_in[10];
    float* out = (float*)d_out;

    // ---- workspace layout (256 B aligned) ----
    char* ws = (char*)d_ws;
    size_t o = 0;
    auto alloc = [&](size_t bytes) {
        char* p = ws + o;
        o += (bytes + 255) & ~(size_t)255;
        return p;
    };
    _Float16* qh      = (_Float16*)alloc((size_t)LQ * CDIM * 2);
    _Float16* vh      = (_Float16*)alloc((size_t)LQ * CDIM * 2);
    _Float16* Wt_off  = (_Float16*)alloc((size_t)256 * 256 * 2);
    _Float16* Wt_attn = (_Float16*)alloc((size_t)128 * 256 * 2);
    _Float16* Wt_val  = (_Float16*)alloc((size_t)256 * 256 * 2);
    _Float16* Wt_out  = (_Float16*)alloc((size_t)256 * 256 * 2);
    float*    off_f   = (float*)   alloc((size_t)LQ * 256 * 4);
    float*    attn_f  = (float*)   alloc((size_t)LQ * 128 * 4);
    _Float16* vproj   = (_Float16*)alloc((size_t)LQ * CDIM * 2);
    _Float16* aout    = (_Float16*)alloc((size_t)LQ * CDIM * 2);

    const int nQC = LQ * CDIM;
    cast_f16_kernel<<<(nQC + 255) / 256, 256, 0, stream>>>(query, qh, nQC);
    cast_f16_kernel<<<(nQC + 255) / 256, 256, 0, stream>>>(value, vh, nQC);
    cast_transpose_f16_kernel<<<(256 * 256 + 255) / 256, 256, 0, stream>>>(W_off,  Wt_off,  256, 256);
    cast_transpose_f16_kernel<<<(256 * 128 + 255) / 256, 256, 0, stream>>>(W_attn, Wt_attn, 256, 128);
    cast_transpose_f16_kernel<<<(256 * 256 + 255) / 256, 256, 0, stream>>>(W_val,  Wt_val,  256, 256);
    cast_transpose_f16_kernel<<<(256 * 256 + 255) / 256, 256, 0, stream>>>(W_out,  Wt_out,  256, 256);

    const int MB = (LQ + 127) / 128;
    // off = query @ W_off + b_off   (fp32 out)
    gemm_wmma_f16<0><<<dim3(MB, 256 / 64), 256, 0, stream>>>(
        qh, Wt_off, b_off, off_f, nullptr, LQ, 256, 256);
    // attn logits = query @ W_attn + b_attn (fp32 out)
    gemm_wmma_f16<0><<<dim3(MB, 128 / 64), 256, 0, stream>>>(
        qh, Wt_attn, b_attn, attn_f, nullptr, LQ, 128, 256);
    // v = value @ W_val + b_val (f16 out, gather source)
    gemm_wmma_f16<1><<<dim3(MB, 256 / 64), 256, 0, stream>>>(
        vh, Wt_val, b_val, nullptr, vproj, LQ, 256, 256);

    // softmax + bilinear sampling + weighted sum -> aout (f16)
    ms_deform_sample<<<LQ, 256, 0, stream>>>(off_f, attn_f, refp, vproj, aout);

    // final projection -> d_out (fp32)
    gemm_wmma_f16<0><<<dim3(MB, 256 / 64), 256, 0, stream>>>(
        aout, Wt_out, b_out, out, nullptr, LQ, 256, 256);
}
